// AdaContrast_44478681317390
// MI455X (gfx1250) — matrix operand order repaired and checked by reading, not compiled, not run on hardware
//
#include <hip/hip_runtime.h>

#define NQ     2048        // number of query features (N)
#define MBANK  131072      // bank rows (M)
#define DD     256         // feature dim (D)
#define CC     126         // classes (C)
#define KNN    10
#define SPLIT  64          // M-splits in phase 1
#define TILE_COLS (MBANK / SPLIT)   // 2048 columns per workgroup
#define TILE_M 16
#define NWAVE  8
#define BIGF   1e30f

typedef __attribute__((ext_vector_type(16))) __bf16 v16bf;
typedef __attribute__((ext_vector_type(8)))  float  v8f;

union FragBF { v16bf v; unsigned int u[8]; uint4 q[2]; };

__device__ __forceinline__ unsigned short f2bf(float f) {
  unsigned int u = __float_as_uint(f);
  u += 0x7FFFu + ((u >> 16) & 1u);           // round-to-nearest-even
  return (unsigned short)(u >> 16);
}

// ---------------- kernel 0a: features -> bf16 of (-2 * f) ----------------
// (-2 folds the cdist cross term into the WMMA so s = b2 + dot(-2f, b))
__global__ void feat_to_bf16(const float* __restrict__ f, unsigned short* __restrict__ o) {
  int i = blockIdx.x * blockDim.x + threadIdx.x;
  if (i < NQ * DD) o[i] = f2bf(-2.0f * f[i]);
}

// ---------------- kernel 0b: bank -> bf16 + b2 (row sum of squares) -------
// one wave per bank row; lane handles 8 contiguous floats
__global__ __launch_bounds__(256)
void bank_to_bf16_b2(const float* __restrict__ bank,
                     unsigned short* __restrict__ bankbf,
                     float* __restrict__ b2g) {
  const int lane = threadIdx.x & 31;
  const int wave = threadIdx.x >> 5;
  const int row  = blockIdx.x * 8 + wave;
  const float4* src = (const float4*)(bank + (size_t)row * DD) + lane * 2;
  float4 f0 = src[0], f1 = src[1];
  float sq = f0.x*f0.x + f0.y*f0.y + f0.z*f0.z + f0.w*f0.w
           + f1.x*f1.x + f1.y*f1.y + f1.z*f1.z + f1.w*f1.w;
  union { unsigned short h[8]; uint4 q; } pk;
  pk.h[0]=f2bf(f0.x); pk.h[1]=f2bf(f0.y); pk.h[2]=f2bf(f0.z); pk.h[3]=f2bf(f0.w);
  pk.h[4]=f2bf(f1.x); pk.h[5]=f2bf(f1.y); pk.h[6]=f2bf(f1.z); pk.h[7]=f2bf(f1.w);
  *((uint4*)(bankbf + (size_t)row * DD) + lane) = pk.q;
#pragma unroll
  for (int off = 16; off > 0; off >>= 1) sq += __shfl_xor(sq, off, 32);
  if (lane == 0) b2g[row] = sq;
}

// replace-worst top-10 insertion (fully unrolled -> cndmask chains)
#define TOPK_INSERT(v_, i_) do {                                   \
    if ((v_) < worst) {                                            \
      _Pragma("unroll")                                            \
      for (int _j = 0; _j < KNN; ++_j)                             \
        if (_j == worstpos) { tv[_j] = (v_); ti[_j] = (i_); }      \
      worst = tv[0]; worstpos = 0;                                 \
      _Pragma("unroll")                                            \
      for (int _j = 1; _j < KNN; ++_j)                             \
        if (tv[_j] > worst) { worst = tv[_j]; worstpos = _j; }     \
    } } while (0)

// ---------------- kernel 1: fused bf16 WMMA GEMM + per-row partial top-10 --
// grid = (SPLIT, NQ/128); block = 256 (8 waves; wave w owns 16 query rows,
// A fragments resident in VGPRs for the whole 2048-column sweep).
template <bool PRECONV>
__global__ __launch_bounds__(256, 1)
void knn_gemm_topk(const float* __restrict__ bank,
                   const unsigned short* __restrict__ bankbf,
                   const float* __restrict__ b2g,
                   const unsigned short* __restrict__ featbf,
                   float* __restrict__ cand_val,
                   int*   __restrict__ cand_idx) {
  __shared__ unsigned short lds_b[TILE_M * DD];       // 8 KB bf16 bank tile
  __shared__ float b2_part[256];
  __shared__ float b2_lds[TILE_M];
  __shared__ float cscratch[NWAVE][16][17];           // per-wave C transpose

  const int tid  = threadIdx.x;
  const int lane = tid & 31;
  const int wave = tid >> 5;
  const int rowBlock = blockIdx.y * (NWAVE * 16);
  const int colBase  = blockIdx.x * TILE_COLS;
  const unsigned ldsb_off = (unsigned)(uintptr_t)&lds_b[0];

  // ---- preload A fragments for this wave (8 k-steps of 16x32 bf16) ----
  FragBF afrag[8];
  {
    const int m  = lane & 15;
    const int kb = (lane < 16) ? 0 : 8;
    const unsigned short* arow = featbf + (size_t)(rowBlock + wave * 16 + m) * DD;
#pragma unroll
    for (int kk = 0; kk < 8; ++kk) {
      afrag[kk].q[0] = *(const uint4*)(arow + kk * 32 + kb);
      afrag[kk].q[1] = *(const uint4*)(arow + kk * 32 + 16 + kb);
    }
  }

  float tv[KNN]; int ti[KNN];
#pragma unroll
  for (int j = 0; j < KNN; ++j) { tv[j] = BIGF; ti[j] = 0; }
  float worst = BIGF; int worstpos = 0;

  for (int mt = 0; mt < TILE_COLS; mt += TILE_M) {
    const int mBase = colBase + mt;
    __syncthreads();   // protect lds_b / b2 from previous iteration readers

    if (PRECONV) {
      // ---- async-copy 8KB bf16 tile straight into LDS (ASYNCcnt path) ----
      const uint4* gsrc = (const uint4*)(bankbf + (size_t)mBase * DD);  // 512 x 16B
      asm volatile("global_load_async_to_lds_b128 %0, %1, off"
                   :: "v"(ldsb_off + (unsigned)tid * 16u), "v"(gsrc + tid)
                   : "memory");
      asm volatile("global_load_async_to_lds_b128 %0, %1, off"
                   :: "v"(ldsb_off + (unsigned)(tid + 256) * 16u), "v"(gsrc + tid + 256)
                   : "memory");
      if (tid < TILE_M) b2_lds[tid] = b2g[mBase + tid];
      if (mt + TILE_M < TILE_COLS && tid < 128)
        __builtin_prefetch((const char*)(bankbf + (size_t)(mBase + TILE_M) * DD)
                           + (size_t)tid * 64, 0, 1);
      asm volatile("s_wait_asynccnt 0" ::: "memory");
      __syncthreads();
    } else {
      // ---- fallback: stage fp32 -> bf16 + b2 reduction in-kernel ----
      if (mt + TILE_M < TILE_COLS) {
        const char* nxt = (const char*)(bank + (size_t)(mBase + TILE_M + (tid >> 4)) * DD)
                          + (size_t)(tid & 15) * 64;
        __builtin_prefetch(nxt, 0, 1);
      }
      {
        const int r   = tid >> 4;
        const int seg = tid & 15;
        const float4* brow = (const float4*)(bank + (size_t)(mBase + r) * DD) + seg * 4;
        unsigned short* drow = lds_b + r * DD + seg * 16;
        float sq = 0.f;
#pragma unroll
        for (int q = 0; q < 4; ++q) {
          float4 f = brow[q];
          sq += f.x * f.x + f.y * f.y + f.z * f.z + f.w * f.w;
          drow[q * 4 + 0] = f2bf(f.x);
          drow[q * 4 + 1] = f2bf(f.y);
          drow[q * 4 + 2] = f2bf(f.z);
          drow[q * 4 + 3] = f2bf(f.w);
        }
        b2_part[tid] = sq;
      }
      __syncthreads();
      if (tid < TILE_M) {
        float s = 0.f;
#pragma unroll
        for (int q = 0; q < 16; ++q) s += b2_part[tid * 16 + q];
        b2_lds[tid] = s;
      }
      __syncthreads();
    }

    // ---- 16x16 tile: acc starts at b2[col], then 8 WMMAs add -2*f.b ----
    const float b2v = b2_lds[lane & 15];
    v8f acc = {b2v, b2v, b2v, b2v, b2v, b2v, b2v, b2v};
    {
      const int kb2 = (lane < 16) ? 0 : 8;
      const unsigned short* brow2 = lds_b + (lane & 15) * DD;
      FragBF bfr[8];
#pragma unroll
      for (int kk = 0; kk < 8; ++kk) {           // 16 ds_load_b128, no waits between
        const uint4* p = (const uint4*)(brow2 + kk * 32 + kb2);
        bfr[kk].q[0] = p[0];
        bfr[kk].q[1] = p[2];
      }
#pragma unroll
      for (int kk = 0; kk < 8; ++kk)
        acc = __builtin_amdgcn_wmma_f32_16x16x32_bf16(
            false, afrag[kk].v, false, bfr[kk].v, (short)0, acc, false, false);
    }

    // ---- transpose s-values through per-wave LDS scratch, fold into top-10 ----
    {
      const int ccol = lane & 15;
      const int rhi  = (lane < 16) ? 0 : 8;
#pragma unroll
      for (int i = 0; i < 8; ++i) cscratch[wave][i + rhi][ccol] = acc[i];
    }
    {
      const int r  = lane & 15;
      const int c0 = (lane < 16) ? 0 : 8;
#pragma unroll
      for (int j = 0; j < 8; ++j) {
        const int col  = c0 + j;
        const float s  = cscratch[wave][r][col];   // already b2 - 2*dot
        const int gidx = mBase + col;
        TOPK_INSERT(s, gidx);
      }
    }
  }

  // ---- merge partner half-lists (lane r <- lane r+16), write to ws ----
#pragma unroll
  for (int j = 0; j < KNN; ++j) {
    float pv = __shfl(tv[j], lane ^ 16, 32);
    int   pi = __shfl(ti[j], lane ^ 16, 32);
    if (lane < 16) { TOPK_INSERT(pv, pi); }
  }
  if (lane < 16) {
    const int grow = rowBlock + wave * 16 + lane;
    const size_t base = ((size_t)grow * SPLIT + blockIdx.x) * KNN;
#pragma unroll
    for (int j = 0; j < KNN; ++j) {
      cand_val[base + j] = tv[j];
      cand_idx[base + j] = ti[j];
    }
  }
}

// ---------------- kernel 2: merge candidates, gather probs, argmax ----------
__global__ __launch_bounds__(256)
void knn_merge_probs(const float* __restrict__ cand_val,
                     const int*   __restrict__ cand_idx,
                     const float* __restrict__ probs,
                     float* __restrict__ out) {
  const int n   = blockIdx.x;
  const int tid = threadIdx.x;
  const int NC  = SPLIT * KNN;    // 640 candidates

  __shared__ float sval[SPLIT * KNN];
  __shared__ int   sidx[SPLIT * KNN];
  __shared__ float rval[256];
  __shared__ int   rpos[256];
  __shared__ int   chosen[KNN];

  for (int i = tid; i < NC; i += 256) {
    sval[i] = cand_val[(size_t)n * NC + i];
    sidx[i] = cand_idx[(size_t)n * NC + i];
  }
  __syncthreads();

  for (int p = 0; p < KNN; ++p) {
    float best = BIGF; int bpos = 0x7fffffff;
    for (int i = tid; i < NC; i += 256)
      if (sval[i] < best) { best = sval[i]; bpos = i; }
    rval[tid] = best; rpos[tid] = bpos;
    __syncthreads();
    for (int off = 128; off > 0; off >>= 1) {
      if (tid < off) {
        float ov = rval[tid + off]; int op = rpos[tid + off];
        if (ov < rval[tid] || (ov == rval[tid] && op < rpos[tid])) {
          rval[tid] = ov; rpos[tid] = op;
        }
      }
      __syncthreads();
    }
    if (tid == 0) { chosen[p] = sidx[rpos[0]]; sval[rpos[0]] = BIGF; }
    __syncthreads();
  }

  float acc = 0.f;
  if (tid < CC) {
#pragma unroll
    for (int j = 0; j < KNN; ++j)
      acc += probs[(size_t)chosen[j] * CC + tid];
    acc *= (1.0f / KNN);
    out[NQ + (size_t)n * CC + tid] = acc;
  }
  rval[tid] = (tid < CC) ? acc : -BIGF;
  rpos[tid] = tid;
  __syncthreads();
  for (int off = 128; off > 0; off >>= 1) {
    if (tid < off) {
      float ov = rval[tid + off]; int op = rpos[tid + off];
      if (ov > rval[tid] || (ov == rval[tid] && op < rpos[tid])) {
        rval[tid] = ov; rpos[tid] = op;
      }
    }
    __syncthreads();
  }
  if (tid == 0) out[n] = (float)rpos[0];
}

extern "C" void kernel_launch(void* const* d_in, const int* in_sizes, int n_in,
                              void* d_out, int out_size, void* d_ws, size_t ws_size,
                              hipStream_t stream) {
  const float* features = (const float*)d_in[0];   // (2048, 256)
  const float* bank     = (const float*)d_in[1];   // (131072, 256)
  const float* probs    = (const float*)d_in[2];   // (131072, 126)
  float* out = (float*)d_out;

  char* ws = (char*)d_ws;
  size_t off = 0;
  unsigned short* featbf = (unsigned short*)(ws + off); off += (size_t)NQ * DD * 2;
  float* cand_val = (float*)(ws + off);              off += (size_t)NQ * SPLIT * KNN * 4;
  int*   cand_idx = (int*)(ws + off);                off += (size_t)NQ * SPLIT * KNN * 4;
  const size_t base_need = off;                               // ~11.5 MB
  unsigned short* bankbf = (unsigned short*)(ws + off); off += (size_t)MBANK * DD * 2;
  float* b2g = (float*)(ws + off);                   off += (size_t)MBANK * 4;
  const bool preconv = (ws_size >= off);             // ~79.1 MB total

  feat_to_bf16<<<(NQ * DD) / 256, 256, 0, stream>>>(features, featbf);
  if (preconv) {
    bank_to_bf16_b2<<<MBANK / 8, 256, 0, stream>>>(bank, bankbf, b2g);
    knn_gemm_topk<true><<<dim3(SPLIT, NQ / (NWAVE * 16)), 256, 0, stream>>>(
        bank, bankbf, b2g, featbf, cand_val, cand_idx);
  } else {
    knn_gemm_topk<false><<<dim3(SPLIT, NQ / (NWAVE * 16)), 256, 0, stream>>>(
        bank, bankbf, b2g, featbf, cand_val, cand_idx);
  }
  knn_merge_probs<<<NQ, 256, 0, stream>>>(cand_val, cand_idx, probs, out);
}
